// GatedRNN_34376918237764
// MI455X (gfx1250) — matrix operand order repaired and checked by reading, not compiled, not run on hardware
//
#include <hip/hip_runtime.h>
#include <math.h>

typedef __bf16 bf16;
typedef __attribute__((ext_vector_type(16))) __bf16 bf16x16;
typedef __attribute__((ext_vector_type(8)))  float  f32x8;

#define RNN_B 256
#define RNN_T 512
#define RNN_D 64
#define RNN_H 256

// packed-weight element offsets (bf16 elements) inside d_ws
#define OFF_W1P 0        // 16x2x512   = 16384
#define OFF_U1P 16384    // 16x8x512   = 65536
#define OFF_W2P 81920    // 65536
#define OFF_U2P 147456   // 65536
#define OFF_O1P 212992   // 4x8x512    = 16384
#define OFF_O2P 229376   // 16384
#define OFF_GP  245760   // 16x16x512  = 131072
#define WS_WEIGHT_BYTES 753664  // 376832 bf16 * 2B

union FragU { uint4 q[2]; bf16x16 v; };

__device__ __forceinline__ f32x8 wmma_bf16(bf16x16 a, bf16x16 b, f32x8 c) {
  return __builtin_amdgcn_wmma_f32_16x16x32_bf16(false, a, false, b, (short)0, c,
                                                 false, false);
}

// A-operand fragment from LDS state (row-major bf16, M=row, K along stride).
// 16-bit A layout: lanes 0-15 -> K {0..7, 16..23}; lanes 16-31 -> K {8..15, 24..31}.
__device__ __forceinline__ bf16x16 frag_a_lds(const bf16* s, int row, int stride,
                                              int kbase, int g) {
  FragU u;
  u.q[0] = *(const uint4*)(s + row * stride + kbase + g * 8);
  u.q[1] = *(const uint4*)(s + row * stride + kbase + 16 + g * 8);
  return u.v;
}

// Opaque (non-hoistable) issue of one B fragment: two global_load_b128.
// volatile asm => LICM cannot hoist it out of the t-loop => no spill storm.
// "=&v" early-clobber: dest quads must not alias the address pair.
__device__ __forceinline__ void frag_b_issue(const bf16* p, FragU& u) {
  asm volatile("global_load_b128 %0, %2, off\n\t"
               "global_load_b128 %1, %2, off offset:16"
               : "=&v"(u.q[0]), "=&v"(u.q[1])
               : "v"(p));
}

// Wait instructions returning an opaque zero token. The token is folded into
// the LDS A-fragment address, so ds_load (and the dependent v_wmma) are
// data-ordered after the wait; B fragments were defined by earlier volatile
// asm (volatile program order is preserved). No tied constraints.
// In-order VMEM return: loadcnt<=2 => all but the 2 newest loads completed.
__device__ __forceinline__ int wait2_tok() {
  int t;
  asm volatile("s_wait_loadcnt 0x2\n\t"
               "s_mov_b32 %0, 0"
               : "=s"(t));
  return t;
}
__device__ __forceinline__ int wait0_tok() {
  int t;
  asm volatile("s_wait_loadcnt 0x0\n\t"
               "s_mov_b32 %0, 0"
               : "=s"(t));
  return t;
}

// One GEMM chain over two A sources / weight tiles with a 2-deep software
// pipeline on the B fragments: issue c+1, wait for c, wmma c.
template <int NC1, int NC2>
__device__ __forceinline__ f32x8 gemm_chain(const bf16* s1, int stride1,
                                            const bf16* w1t, const bf16* s2,
                                            const bf16* w2t, int lr, int g,
                                            float binit) {
  constexpr int NC = NC1 + NC2;
  FragU fb[2];
  frag_b_issue(w1t, fb[0]);
  f32x8 acc;
#pragma unroll
  for (int v = 0; v < 8; ++v) acc[v] = binit;
#pragma unroll
  for (int c = 0; c < NC; ++c) {
    int tok;
    if (c + 1 < NC) {
      const bf16* pn = (c + 1 < NC1) ? (w1t + (size_t)(c + 1) * 512)
                                     : (w2t + (size_t)(c + 1 - NC1) * 512);
      frag_b_issue(pn, fb[(c + 1) & 1]);
      tok = wait2_tok();
    } else {
      tok = wait0_tok();
    }
    bf16x16 a = (c < NC1)
                    ? frag_a_lds(s1, lr, stride1, c * 32 + tok, g)
                    : frag_a_lds(s2, lr, RNN_H, (c - NC1) * 32 + tok, g);
    acc = wmma_bf16(a, fb[c & 1].v, acc);
  }
  return acc;
}

// Pack W (N_out x Kdim, row-major fp32) into WMMA B-fragment-major bf16.
// B[k][N] = W[N][k]; lane: g=lane/16 selects K half, lane%16 = column.
__global__ void rnn_pack(const float* __restrict__ src, bf16* __restrict__ dst,
                         int Kdim, int numK, int total) {
  int p = blockIdx.x * blockDim.x + threadIdx.x;
  if (p >= total) return;
  int per_n = numK * 512;
  int n  = p / per_n;
  int r  = p - n * per_n;
  int c  = r >> 9;
  int q  = r & 511;
  int ln = q >> 4;
  int i  = q & 15;
  int gg = ln >> 4;
  int cc = ln & 15;
  int k  = c * 32 + gg * 8 + (i & 7) + ((i >> 3) << 4);
  int N  = n * 16 + cc;
  dst[p] = (bf16)src[(size_t)N * Kdim + k];
}

__global__ void rnn_bias(const float* W1b, const float* U1b, const float* W2b,
                         const float* U2b, const float* o1b, const float* o2b,
                         const float* gb, float* bias) {
  int i = threadIdx.x;                 // 256 threads
  bias[i]       = W1b[i] + U1b[i];     // b1
  bias[256 + i] = W2b[i] + U2b[i];     // b2
  bias[576 + i] = gb[i];               // bg
  if (i < 64) bias[512 + i] = o1b[i] + o2b[i];  // bo
}

__global__ __launch_bounds__(512, 1)
void rnn_main(const float* __restrict__ x, const bf16* __restrict__ wp,
              const float* __restrict__ bias, float* __restrict__ out) {
  __shared__ bf16 h1s[16 * RNN_H];
  __shared__ bf16 h2s[16 * RNN_H];
  __shared__ bf16 xs[16 * RNN_D];

  const int tid  = threadIdx.x;
  const int wave = tid >> 5;
  const int lane = tid & 31;
  const int g    = lane >> 4;
  const int lr   = lane & 15;
  const int wg   = blockIdx.x;
  const int ncol = wave * 16 + lr;

  // per-wave N-tile weight pointers (fragment-major; chunk stride 512 elems)
  const size_t lo = (size_t)lane * 16;
  const bf16* W1t = wp + OFF_W1P + (size_t)wave * 2 * 512 + lo;
  const bf16* U1t = wp + OFF_U1P + (size_t)wave * 8 * 512 + lo;
  const bf16* W2t = wp + OFF_W2P + (size_t)wave * 8 * 512 + lo;
  const bf16* U2t = wp + OFF_U2P + (size_t)wave * 8 * 512 + lo;
  const bf16* o1t = wp + OFF_O1P + (size_t)wave * 8 * 512 + lo;   // waves 0..3
  const bf16* o2t = wp + OFF_O2P + (size_t)wave * 8 * 512 + lo;   // waves 0..3
  const bf16* g1t = wp + OFF_GP  + (size_t)wave * 16 * 512 + lo;  // chunks 0..7
  const bf16* g2t = g1t + 8 * 512;                                 // chunks 8..15

  const float bias1 = bias[ncol];
  const float bias2 = bias[256 + ncol];
  const float biaso = (wave < 4) ? bias[512 + ncol] : 0.0f;
  const float biasg = bias[576 + ncol];

  for (int i = tid; i < 16 * RNN_H; i += 512) {
    h1s[i] = (bf16)0.0f;
    h2s[i] = (bf16)0.0f;
  }

  float h2r[8];  // fp32 h2 state tile (wave-local columns)
  float ur[8];   // gate state, initial u = 1
#pragma unroll
  for (int v = 0; v < 8; ++v) { h2r[v] = 0.0f; ur[v] = 1.0f; }

  __syncthreads();

  for (int t = 0; t < RNN_T; ++t) {
    // stage x_t batch tile to LDS (fp32 -> bf16), 2 elems / thread
    {
      const int e = tid << 1;
      const int r = e >> 6;
      const int d = e & 63;
      const float2 xv =
          *(const float2*)(x + ((size_t)(wg * 16 + r) * RNN_T + t) * RNN_D + d);
      xs[r * RNN_D + d]     = (bf16)xv.x;
      xs[r * RNN_D + d + 1] = (bf16)xv.y;
    }
    __syncthreads();

    // h1 = tanh(xt@W1^T + h1@U1^T + b1), K = 64 + 256
    f32x8 acc = gemm_chain<2, 8>(xs, RNN_D, W1t, h1s, U1t, lr, g, bias1);
    float h1v[8];
#pragma unroll
    for (int v = 0; v < 8; ++v) h1v[v] = tanhf(acc[v]);

    __syncthreads();  // all reads of old h1 complete
#pragma unroll
    for (int v = 0; v < 8; ++v)
      h1s[(v + 8 * g) * RNN_H + ncol] = (bf16)h1v[v];
    __syncthreads();  // new h1 visible

    // h2_new = tanh(h1@W2^T + h2@U2^T + b2); h2 = u*h2_new + (1-u)*h2
    f32x8 acc2 = gemm_chain<8, 8>(h1s, RNN_H, W2t, h2s, U2t, lr, g, bias2);
#pragma unroll
    for (int v = 0; v < 8; ++v) {
      const float hn = tanhf(acc2[v]);
      h2r[v] = ur[v] * hn + (1.0f - ur[v]) * h2r[v];
    }
    __syncthreads();  // all reads of old h2 complete
#pragma unroll
    for (int v = 0; v < 8; ++v)
      h2s[(v + 8 * g) * RNN_H + ncol] = (bf16)h2r[v];
    __syncthreads();  // new h2 visible

    // out = h1@o1^T + h2@o2^T + bo   (D=64 -> 4 N-tiles, waves 0..3)
    if (wave < 4) {
      f32x8 acco = gemm_chain<8, 8>(h1s, RNN_H, o1t, h2s, o2t, lr, g, biaso);
#pragma unroll
      for (int v = 0; v < 8; ++v)
        out[((size_t)(wg * 16 + v + 8 * g) * RNN_T + t) * RNN_D + ncol] = acco[v];
    }

    // u = sigmoid([h1,h2]@g^T + bg) for the next step; K = 512
    f32x8 accg = gemm_chain<8, 8>(h1s, RNN_H, g1t, h2s, g2t, lr, g, biasg);
#pragma unroll
    for (int v = 0; v < 8; ++v)
      ur[v] = 1.0f / (1.0f + __expf(-accg[v]));
  }
}

extern "C" void kernel_launch(void* const* d_in, const int* in_sizes, int n_in,
                              void* d_out, int out_size, void* d_ws, size_t ws_size,
                              hipStream_t stream) {
  const float* x    = (const float*)d_in[0];
  const float* W1_w = (const float*)d_in[1];
  const float* W1_b = (const float*)d_in[2];
  const float* U1_w = (const float*)d_in[3];
  const float* U1_b = (const float*)d_in[4];
  const float* W2_w = (const float*)d_in[5];
  const float* W2_b = (const float*)d_in[6];
  const float* U2_w = (const float*)d_in[7];
  const float* U2_b = (const float*)d_in[8];
  const float* o1_w = (const float*)d_in[9];
  const float* o1_b = (const float*)d_in[10];
  const float* o2_w = (const float*)d_in[11];
  const float* o2_b = (const float*)d_in[12];
  const float* g_w  = (const float*)d_in[13];
  const float* g_b  = (const float*)d_in[14];

  bf16*  wsb  = (bf16*)d_ws;
  float* bptr = (float*)((char*)d_ws + WS_WEIGHT_BYTES);

  // pack weights into WMMA B-fragment layout (bf16)
  auto pack = [&](const float* src, bf16* dst, int Kdim, int numK, int total) {
    rnn_pack<<<(total + 255) / 256, 256, 0, stream>>>(src, dst, Kdim, numK, total);
  };
  pack(W1_w, wsb + OFF_W1P, RNN_D, 2, 16384);
  pack(U1_w, wsb + OFF_U1P, RNN_H, 8, 65536);
  pack(W2_w, wsb + OFF_W2P, RNN_H, 8, 65536);
  pack(U2_w, wsb + OFF_U2P, RNN_H, 8, 65536);
  pack(o1_w, wsb + OFF_O1P, RNN_H, 8, 16384);
  pack(o2_w, wsb + OFF_O2P, RNN_H, 8, 16384);
  pack(g_w,  wsb + OFF_GP,  2 * RNN_H, 16, 131072);

  rnn_bias<<<1, 256, 0, stream>>>(W1_b, U1_b, W2_b, U2_b, o1_b, o2_b, g_b, bptr);

  // persistent recurrence: 16 batch tiles x 16 waves (one WGP workgroup each)
  rnn_main<<<RNN_B / 16, 512, 0, stream>>>(x, wsb, bptr, (float*)d_out);
}
